// Attn_block_46712064311824
// MI455X (gfx1250) — matrix operand order repaired and checked by reading, not compiled, not run on hardware
//
#include <hip/hip_runtime.h>
#include <stdint.h>

#define B_  2
#define C_  1024
#define L_  2048
#define H_  16
#define HD_ 64
#define BH_ (B_*H_)

typedef __bf16 bf16_t;
typedef __attribute__((ext_vector_type(16))) __bf16 v16bf;
typedef __attribute__((ext_vector_type(8)))  float  v8f;

union BFrag { v16bf v; uint4 q[2]; __bf16 h[16]; };

__device__ __forceinline__ v8f wmma_bf16(const BFrag& a, const BFrag& b, v8f c) {
  // D = A(16x32 bf16) * B(32x16 bf16) + C(16x16 f32)
  return __builtin_amdgcn_wmma_f32_16x16x32_bf16(false, a.v, false, b.v,
                                                 (short)0, c, false, false);
}

// A fragment: lane holds row M=(lane&15); elems 0..15 = K[k0+half*16 .. +15]
__device__ __forceinline__ BFrag load_a(const bf16_t* row, int k0, int half) {
  BFrag f;
  f.q[0] = *(const uint4*)(row + k0 + half * 16);
  f.q[1] = *(const uint4*)(row + k0 + half * 16 + 8);
  return f;
}
// B fragment: lane holds col N=(lane&15); elems 0..15 = K[half*16 .. +15]
__device__ __forceinline__ BFrag load_b16(const bf16_t* p) {
  BFrag f;
  f.q[0] = *(const uint4*)(p);
  f.q[1] = *(const uint4*)(p + 8);
  return f;
}
__device__ __forceinline__ v8f vzero8() { v8f z = {0,0,0,0,0,0,0,0}; return z; }

// ---------------- conversion kernels ----------------
__global__ void cvt_w_kernel(const float* __restrict__ src, bf16_t* __restrict__ dst, int n) {
  int i = blockIdx.x * blockDim.x + threadIdx.x;
  if (i < n) dst[i] = (__bf16)src[i];
}

__global__ void cvt_x_kernel(const float* __restrict__ x,
                             bf16_t* __restrict__ xbf,   // [B,C,L]
                             bf16_t* __restrict__ xt) {  // [B,L,C]
  int i = blockIdx.x * blockDim.x + threadIdx.x;
  if (i >= B_ * C_ * L_) return;
  int l = i % L_;
  int c = (i / L_) % C_;
  int b = i / (L_ * C_);
  float v = x[i];
  xbf[i] = (__bf16)v;
  xt[((size_t)b * L_ + l) * C_ + c] = (__bf16)v;
}

// ---------------- 1x1-conv GEMM: Out[o,l] = sum_c W[o,c] * In[c,l] ----------------
// W: [C,C] bf16 row-major;  Tin: [B,L,C] bf16 (transposed input -> contiguous K)
// Wave computes a 64(M) x 16(N) tile; B-fragment shared across 4 M-subtiles.
__global__ __launch_bounds__(256) void gemm_conv(
    const bf16_t* __restrict__ W, const float* __restrict__ bias,
    const bf16_t* __restrict__ Tin,
    const float* __restrict__ res1, const float* __restrict__ res2, // [B,C,L] or null
    float*  __restrict__ outF,   // [B,C,L] fp32 or null
    bf16_t* __restrict__ outT,   // [B,L,C] bf16 or null
    int do_relu)
{
  const int lane = threadIdx.x & 31;
  const int wave = threadIdx.x >> 5;
  const int half = lane >> 4;
  const int l15  = lane & 15;
  const int b    = blockIdx.z;
  const int lb   = blockIdx.x * 16;
  const int ob   = blockIdx.y * 512 + wave * 64;
  const int lcol = lb + l15;

  v8f acc[4];
  #pragma unroll
  for (int t = 0; t < 4; ++t) acc[t] = vzero8();

  const bf16_t* tin_row = Tin + ((size_t)b * L_ + lcol) * C_;
  for (int k0 = 0; k0 < C_; k0 += 32) {
    BFrag bf = load_b16(tin_row + k0 + half * 16);
    #pragma unroll
    for (int t = 0; t < 4; ++t) {
      const bf16_t* wrow = W + (size_t)(ob + t * 16 + l15) * C_;
      BFrag af = load_a(wrow, k0, half);
      acc[t] = wmma_bf16(af, bf, acc[t]);
    }
  }

  #pragma unroll
  for (int t = 0; t < 4; ++t) {
    float vals[8];
    #pragma unroll
    for (int r = 0; r < 8; ++r) {
      int row = ob + t * 16 + r + half * 8;           // D: lanes16-31 hold M=r+8
      float v = acc[t][r] + bias[row];
      size_t idx = ((size_t)b * C_ + row) * L_ + lcol;
      if (res1) v += res1[idx];
      if (res2) v += res2[idx];
      if (do_relu) v = v > 0.f ? v : 0.f;
      vals[r] = v;
      if (outF) outF[idx] = v;
    }
    if (outT) {
      union { uint4 q; __bf16 h[8]; } pk;
      #pragma unroll
      for (int r = 0; r < 8; ++r) pk.h[r] = (__bf16)vals[r];
      *(uint4*)(outT + ((size_t)b * L_ + lcol) * C_ + ob + t * 16 + half * 8) = pk.q;
    }
  }
}

// ---------------- attention pass 1: per-row (l) softmax stats ----------------
// scores[l,m] = sum_c Kt[l,c]*Qt[m,c]; stats[bh,l] = (rowmax, rowsumexp) of scores*scale
__global__ __launch_bounds__(256) void attn_stats(
    const bf16_t* __restrict__ Kt, const bf16_t* __restrict__ Qt,  // [B,L,C]
    float2* __restrict__ stats)                                    // [BH,L]
{
  const int lane = threadIdx.x & 31;
  const int wave = threadIdx.x >> 5;
  const int half = lane >> 4;
  const int l15  = lane & 15;
  const int bh   = blockIdx.y;
  const int b    = bh / H_;
  const int h    = bh % H_;
  const int lb   = (blockIdx.x * 8 + wave) * 16;
  const float scale = 2.0f / (float)L_;   // softmax(scores / (L/2))
  const size_t coff = (size_t)h * HD_;

  const bf16_t* krow = Kt + ((size_t)b * L_ + lb + l15) * C_ + coff;
  BFrag a0 = load_a(krow, 0, half);
  BFrag a1 = load_a(krow, 32, half);

  float m_run[8], s_run[8];
  #pragma unroll
  for (int r = 0; r < 8; ++r) { m_run[r] = -1e30f; s_run[r] = 0.f; }

  for (int mb = 0; mb < L_; mb += 16) {
    const bf16_t* qrow = Qt + ((size_t)b * L_ + mb + l15) * C_ + coff;
    BFrag b0 = load_b16(qrow + half * 16);
    BFrag b1 = load_b16(qrow + 32 + half * 16);
    v8f s = vzero8();
    s = wmma_bf16(a0, b0, s);
    s = wmma_bf16(a1, b1, s);
    #pragma unroll
    for (int r = 0; r < 8; ++r) {
      float v = s[r] * scale;
      float tmax = v;                       // reduce over the 16 lanes of this half
      tmax = fmaxf(tmax, __shfl_xor(tmax, 1, 32));
      tmax = fmaxf(tmax, __shfl_xor(tmax, 2, 32));
      tmax = fmaxf(tmax, __shfl_xor(tmax, 4, 32));
      tmax = fmaxf(tmax, __shfl_xor(tmax, 8, 32));
      float newm = fmaxf(m_run[r], tmax);
      float p = __expf(v - newm);
      float tsum = p;
      tsum += __shfl_xor(tsum, 1, 32);
      tsum += __shfl_xor(tsum, 2, 32);
      tsum += __shfl_xor(tsum, 4, 32);
      tsum += __shfl_xor(tsum, 8, 32);
      s_run[r] = s_run[r] * __expf(m_run[r] - newm) + tsum;
      m_run[r] = newm;
    }
  }
  if (l15 == 0) {
    #pragma unroll
    for (int r = 0; r < 8; ++r) {
      int l = lb + r + half * 8;
      stats[(size_t)bh * L_ + l] = make_float2(m_run[r], s_run[r]);
    }
  }
}

// ---------------- attention pass 2: At[m,c] = sum_l attn[l,m] * X[c,l] ----------------
// Score D-fragments (s0: l in [lb,lb+16), s1: [lb+16,lb+32)) map in-register onto
// the A-fragment (K=l) of the outer 16x16x32 WMMA — exp/scale applied in place.
__global__ __launch_bounds__(256) void attn_out(
    const bf16_t* __restrict__ Kt, const bf16_t* __restrict__ Qt,  // [B,L,C]
    const bf16_t* __restrict__ Xbf,                                // [B,C,L]
    const float2* __restrict__ stats,                              // [BH,L]
    bf16_t* __restrict__ At)                                       // [B,L,C]
{
  const int lane = threadIdx.x & 31;
  const int wave = threadIdx.x >> 5;
  const int half = lane >> 4;
  const int l15  = lane & 15;
  const int bh   = blockIdx.y;
  const int b    = bh / H_;
  const int h    = bh % H_;
  const int mb   = (blockIdx.x * 8 + wave) * 16;
  const float scale = 2.0f / (float)L_;
  const size_t coff = (size_t)h * HD_;

  const bf16_t* qrow = Qt + ((size_t)b * L_ + mb + l15) * C_ + coff;
  BFrag bq0 = load_b16(qrow + half * 16);        // hoisted: query tile fixed
  BFrag bq1 = load_b16(qrow + 32 + half * 16);

  v8f acc[4];
  #pragma unroll
  for (int t = 0; t < 4; ++t) acc[t] = vzero8();
  const float2* st = stats + (size_t)bh * L_;

  for (int lbv = 0; lbv < L_; lbv += 32) {
    const bf16_t* k0row = Kt + ((size_t)b * L_ + lbv + l15) * C_ + coff;
    const bf16_t* k1row = k0row + (size_t)16 * C_;
    BFrag a00 = load_a(k0row, 0, half), a01 = load_a(k0row, 32, half);
    BFrag a10 = load_a(k1row, 0, half), a11 = load_a(k1row, 32, half);
    v8f s0 = vzero8(), s1 = vzero8();
    s0 = wmma_bf16(a00, bq0, s0);  s0 = wmma_bf16(a01, bq1, s0);
    s1 = wmma_bf16(a10, bq0, s1);  s1 = wmma_bf16(a11, bq1, s1);

    BFrag ap;                                   // P^T fragment, K = l
    #pragma unroll
    for (int r = 0; r < 8; ++r) {
      int l0 = lbv + r + half * 8;
      float2 t0 = st[l0];
      float2 t1 = st[l0 + 16];
      float p0 = __expf(s0[r] * scale - t0.x) / t0.y;
      float p1 = __expf(s1[r] * scale - t1.x) / t1.y;
      ap.h[r]     = (__bf16)p0;                 // K = lbv + {0..7|8..15}  (by half)
      ap.h[r + 8] = (__bf16)p1;                 // K = lbv + {16..23|24..31}
    }
    #pragma unroll
    for (int ct = 0; ct < 4; ++ct) {
      const bf16_t* xrow = Xbf + ((size_t)b * C_ + coff + ct * 16 + l15) * L_
                               + lbv + half * 8;
      BFrag bx; bx.q[0] = *(const uint4*)(xrow); bx.q[1] = *(const uint4*)(xrow + 16);
      acc[ct] = wmma_bf16(ap, bx, acc[ct]);
    }
  }

  #pragma unroll
  for (int ct = 0; ct < 4; ++ct)
    #pragma unroll
    for (int r = 0; r < 8; ++r) {
      int m = mb + r + half * 8;
      int c = (int)coff + ct * 16 + l15;
      At[((size_t)b * L_ + m) * C_ + c] = (__bf16)acc[ct][r];
    }
}

// ---------------- host ----------------
extern "C" void kernel_launch(void* const* d_in, const int* in_sizes, int n_in,
                              void* d_out, int out_size, void* d_ws, size_t ws_size,
                              hipStream_t stream) {
  const float* x   = (const float*)d_in[0];
  const float* kw  = (const float*)d_in[1];
  const float* kb  = (const float*)d_in[2];
  const float* qw  = (const float*)d_in[3];
  const float* qb  = (const float*)d_in[4];
  const float* pw  = (const float*)d_in[5];
  const float* pb  = (const float*)d_in[6];
  const float* c1w = (const float*)d_in[7];
  const float* c1b = (const float*)d_in[8];
  const float* c2w = (const float*)d_in[9];
  const float* c2b = (const float*)d_in[10];

  const size_t nBCL = (size_t)B_ * C_ * L_;
  const size_t nCC  = (size_t)C_ * C_;

  char* ws = (char*)d_ws;
  size_t off = 0;
  auto carve = [&](size_t bytes) -> void* {
    void* p = ws + off;
    off += (bytes + 255) & ~(size_t)255;
    return p;
  };
  bf16_t* xbf   = (bf16_t*)carve(nBCL * 2);   // [B,C,L]
  bf16_t* xt    = (bf16_t*)carve(nBCL * 2);   // [B,L,C]
  bf16_t* kwbf  = (bf16_t*)carve(nCC * 2);
  bf16_t* qwbf  = (bf16_t*)carve(nCC * 2);
  bf16_t* pwbf  = (bf16_t*)carve(nCC * 2);
  bf16_t* c1wbf = (bf16_t*)carve(nCC * 2);
  bf16_t* c2wbf = (bf16_t*)carve(nCC * 2);
  bf16_t* kt    = (bf16_t*)carve(nBCL * 2);   // keys^T   [B,L,C]
  bf16_t* qt    = (bf16_t*)carve(nBCL * 2);   // queries^T
  bf16_t* at    = (bf16_t*)carve(nBCL * 2);   // attn out^T
  float*  y     = (float*) carve(nBCL * 4);   // y = p(attn)+x, fp32 [B,C,L]
  bf16_t* yt    = (bf16_t*)carve(nBCL * 2);   // y^T
  bf16_t* rt    = (bf16_t*)carve(nBCL * 2);   // relu(c1(y))^T
  float2* stats = (float2*)carve((size_t)BH_ * L_ * sizeof(float2));

  // 1) precision conversions
  cvt_x_kernel<<<dim3((unsigned)((nBCL + 255) / 256)), 256, 0, stream>>>(x, xbf, xt);
  cvt_w_kernel<<<dim3((unsigned)((nCC + 255) / 256)), 256, 0, stream>>>(kw,  kwbf,  (int)nCC);
  cvt_w_kernel<<<dim3((unsigned)((nCC + 255) / 256)), 256, 0, stream>>>(qw,  qwbf,  (int)nCC);
  cvt_w_kernel<<<dim3((unsigned)((nCC + 255) / 256)), 256, 0, stream>>>(pw,  pwbf,  (int)nCC);
  cvt_w_kernel<<<dim3((unsigned)((nCC + 255) / 256)), 256, 0, stream>>>(c1w, c1wbf, (int)nCC);
  cvt_w_kernel<<<dim3((unsigned)((nCC + 255) / 256)), 256, 0, stream>>>(c2w, c2wbf, (int)nCC);

  const dim3 ggrid(L_ / 16, C_ / 512, B_);    // 16N x 512M tiles per block x batch
  // 2) k/q projections -> transposed bf16
  gemm_conv<<<ggrid, 256, 0, stream>>>(kwbf, kb, xt, nullptr, nullptr, nullptr, kt, 0);
  gemm_conv<<<ggrid, 256, 0, stream>>>(qwbf, qb, xt, nullptr, nullptr, nullptr, qt, 0);
  // 3) softmax row stats, then attn * V (two-pass, scores never hit memory)
  attn_stats<<<dim3(L_ / 128, BH_), 256, 0, stream>>>(kt, qt, stats);
  attn_out  <<<dim3(L_ / 128, BH_), 256, 0, stream>>>(kt, qt, xbf, stats, at);
  // 4) output projection + residual:  y = p(at) + pb + x
  gemm_conv<<<ggrid, 256, 0, stream>>>(pwbf, pb, at, x, nullptr, y, yt, 0);
  // 5) FFN:  r = relu(c1(y) + c1b)
  gemm_conv<<<ggrid, 256, 0, stream>>>(c1wbf, c1b, yt, nullptr, nullptr, nullptr, rt, 1);
  // 6) out = c2(r) + c2b + y + x
  gemm_conv<<<ggrid, 256, 0, stream>>>(c2wbf, c2b, rt, y, x, (float*)d_out, nullptr, 0);
}
